// DioraBase_79388175499443
// MI455X (gfx1250) — matrix operand order repaired and checked by reading, not compiled, not run on hardware
//
#include <hip/hip_runtime.h>
#include <hip/hip_bf16.h>

typedef __attribute__((ext_vector_type(16))) __bf16 v16bf;
typedef __attribute__((ext_vector_type(8)))  __bf16 v8bf;
typedef __attribute__((ext_vector_type(8)))  float  v8f;

#define BB 48
#define TT 32
#define DD 1024
#define SS 512
#define NCELLS 528   // 32*33/2

// offsets[n] for T=32: sum_{i<n} (32-i) = 32n - n(n-1)/2
__device__ __forceinline__ int cell_off(int n) { return n * TT - (n * (n - 1)) / 2; }

__device__ __forceinline__ __bf16 f2bf(float f) {
    unsigned u = __builtin_bit_cast(unsigned, f);
    unsigned r = 0x7FFFu + ((u >> 16) & 1u);
    u += r;
    unsigned short h = (unsigned short)(u >> 16);
    return __builtin_bit_cast(__bf16, h);
}

// A operand (16x32 bf16): lane = 16*half + r holds row r.
// elements 0..7  -> K = kb + half*8 + {0..7}
// elements 8..15 -> K = kb + 16 + half*8 + {0..7}
__device__ __forceinline__ v16bf load_a_tile(const __bf16* rowptr, int half) {
    v8bf lo = *reinterpret_cast<const v8bf*>(rowptr + half * 8);
    v8bf hi = *reinterpret_cast<const v8bf*>(rowptr + 16 + half * 8);
    v16bf a;
#pragma unroll
    for (int i = 0; i < 8; ++i) { a[i] = lo[i]; a[i + 8] = hi[i]; }
    return a;
}

// B operand (32x16 bf16): lane = 16*half + q holds column q.
// elements 0..7 -> K = kb + half*8 + {0..7}, 8..15 -> K = kb + 16 + half*8 + {0..7} ; colptr = wT[col] + kb
__device__ __forceinline__ v16bf load_b_tile(const __bf16* colptr, int half) {
    v8bf lo = *reinterpret_cast<const v8bf*>(colptr + half * 8);
    v8bf hi = *reinterpret_cast<const v8bf*>(colptr + 16 + half * 8);
    v16bf b;
#pragma unroll
    for (int i = 0; i < 8; ++i) { b[i] = lo[i]; b[i + 8] = hi[i]; }
    return b;
}

__device__ __forceinline__ v8f wmma_bf16(v16bf a, v16bf b, v8f c) {
    return __builtin_amdgcn_wmma_f32_16x16x32_bf16(false, a, false, b, (short)0, c, false, false);
}

// ---------------- conversion helpers ----------------

__global__ void cvt_x_kernel(const float* __restrict__ x, __bf16* __restrict__ xb, int n) {
    int i = blockIdx.x * blockDim.x + threadIdx.x;
    if (i < n) xb[i] = f2bf(x[i]);
}

// src: K x N row-major fp32  ->  dst: N x K bf16
__global__ void transpose_w_kernel(const float* __restrict__ w, __bf16* __restrict__ wT,
                                   int K, int N) {
    int i = blockIdx.x * blockDim.x + threadIdx.x;
    if (i < K * N) {
        int ncol = i / K, k = i % K;
        wT[i] = f2bf(w[(size_t)k * N + ncol]);
    }
}

__global__ void zero_f32_kernel(float* __restrict__ p, int n) {
    int i = blockIdx.x * blockDim.x + threadIdx.x;
    if (i < n) p[i] = 0.f;
}

// ---------------- leaf: unit(relu(x @ w_leaf + b)) ----------------
// grid = (B*T)/16 blocks, 256 threads (8 waves x 64 cols)
__global__ __launch_bounds__(256)
void leaf_kernel(const __bf16* __restrict__ xb, const __bf16* __restrict__ wT,
                 const float* __restrict__ bias, float* chart_h, __bf16* chart_hb) {
    const int lane = threadIdx.x & 31;
    const int wave = threadIdx.x >> 5;
    const int half = lane >> 4;
    const int q    = lane & 15;
    const int r0   = blockIdx.x * 16;
    const int colbase = wave * 64;

    v8f acc[4] = {};
    const __bf16* arow  = xb + (size_t)(r0 + q) * DD;
    const __bf16* bbase = wT + (size_t)(colbase + q) * DD;

    // software pipeline: preload K-step 0
    v16bf a  = load_a_tile(arow, half);
    v16bf b0 = load_b_tile(bbase + 0 * 16 * DD, half);
    v16bf b1 = load_b_tile(bbase + 1 * 16 * DD, half);
    v16bf b2 = load_b_tile(bbase + 2 * 16 * DD, half);
    v16bf b3 = load_b_tile(bbase + 3 * 16 * DD, half);
    for (int kb = 32; kb < DD; kb += 32) {
        v16bf an  = load_a_tile(arow + kb, half);
        v16bf bn0 = load_b_tile(bbase + 0 * 16 * DD + kb, half);
        v16bf bn1 = load_b_tile(bbase + 1 * 16 * DD + kb, half);
        v16bf bn2 = load_b_tile(bbase + 2 * 16 * DD + kb, half);
        v16bf bn3 = load_b_tile(bbase + 3 * 16 * DD + kb, half);
        acc[0] = wmma_bf16(a, b0, acc[0]);
        acc[1] = wmma_bf16(a, b1, acc[1]);
        acc[2] = wmma_bf16(a, b2, acc[2]);
        acc[3] = wmma_bf16(a, b3, acc[3]);
        a = an; b0 = bn0; b1 = bn1; b2 = bn2; b3 = bn3;
    }
    acc[0] = wmma_bf16(a, b0, acc[0]);
    acc[1] = wmma_bf16(a, b1, acc[1]);
    acc[2] = wmma_bf16(a, b2, acc[2]);
    acc[3] = wmma_bf16(a, b3, acc[3]);

    __shared__ float red[16];
    if (threadIdx.x < 16) red[threadIdx.x] = 0.f;
    __syncthreads();

    float ss[8];
#pragma unroll
    for (int j = 0; j < 8; ++j) ss[j] = 0.f;
#pragma unroll
    for (int nt = 0; nt < 4; ++nt) {
        float bv = bias[colbase + nt * 16 + q];
#pragma unroll
        for (int j = 0; j < 8; ++j) {
            float v = acc[nt][j] + bv;
            v = v > 0.f ? v : 0.f;
            acc[nt][j] = v;
            ss[j] += v * v;
        }
    }
#pragma unroll
    for (int j = 0; j < 8; ++j) {
        float v = ss[j];
        v += __shfl_xor(v, 1); v += __shfl_xor(v, 2);
        v += __shfl_xor(v, 4); v += __shfl_xor(v, 8);
        ss[j] = v;
    }
    if (q == 0) {
#pragma unroll
        for (int j = 0; j < 8; ++j) atomicAdd(&red[j + 8 * half], ss[j]);
    }
    __syncthreads();

#pragma unroll
    for (int j = 0; j < 8; ++j) {
        int m   = j + 8 * half;
        int row = r0 + m;
        float inv = 1.f / fmaxf(sqrtf(red[m]), 1e-12f);
        int b = row >> 5, t = row & 31;
        size_t base = ((size_t)b * NCELLS + t) * SS;
#pragma unroll
        for (int nt = 0; nt < 4; ++nt) {
            int col = colbase + nt * 16 + q;
            float v = acc[nt][j] * inv;
            chart_h[base + col]  = v;
            chart_hb[base + col] = f2bf(v);
        }
    }
}

// ---------------- score: s = lh . (S . rh) + ls + rs ----------------
// rows = B*L*N flattened as b*(L*N) + pos*N + n ; grid = ceil(rows/16) x 256
__global__ __launch_bounds__(256)
void score_kernel(const __bf16* __restrict__ chart_hb, const float* __restrict__ chart_h,
                  const __bf16* __restrict__ sT, const float* __restrict__ chart_s,
                  float* __restrict__ sbuf, int level, int L, int rows) {
    const int lane = threadIdx.x & 31;
    const int wave = threadIdx.x >> 5;
    const int half = lane >> 4;
    const int q    = lane & 15;
    const int r0   = blockIdx.x * 16;
    const int N    = level;
    const int LN   = L * N;
    const int colbase = wave * 64;

    int r = r0 + q;
    int rb = 0, pos = 0, n = 0;
    if (r < rows) { rb = r / LN; int rem = r % LN; pos = rem / N; n = rem % N; }
    int lhc = cell_off(n) + pos;
    const __bf16* arow  = chart_hb + ((size_t)rb * NCELLS + lhc) * SS;
    const __bf16* bbase = sT + (size_t)(colbase + q) * SS;

    v8f acc[4] = {};
    v16bf a  = load_a_tile(arow, half);
    v16bf b0 = load_b_tile(bbase + 0 * 16 * SS, half);
    v16bf b1 = load_b_tile(bbase + 1 * 16 * SS, half);
    v16bf b2 = load_b_tile(bbase + 2 * 16 * SS, half);
    v16bf b3 = load_b_tile(bbase + 3 * 16 * SS, half);
    for (int kb = 32; kb < SS; kb += 32) {
        v16bf an  = load_a_tile(arow + kb, half);
        v16bf bn0 = load_b_tile(bbase + 0 * 16 * SS + kb, half);
        v16bf bn1 = load_b_tile(bbase + 1 * 16 * SS + kb, half);
        v16bf bn2 = load_b_tile(bbase + 2 * 16 * SS + kb, half);
        v16bf bn3 = load_b_tile(bbase + 3 * 16 * SS + kb, half);
        acc[0] = wmma_bf16(a, b0, acc[0]);
        acc[1] = wmma_bf16(a, b1, acc[1]);
        acc[2] = wmma_bf16(a, b2, acc[2]);
        acc[3] = wmma_bf16(a, b3, acc[3]);
        a = an; b0 = bn0; b1 = bn1; b2 = bn2; b3 = bn3;
    }
    acc[0] = wmma_bf16(a, b0, acc[0]);
    acc[1] = wmma_bf16(a, b1, acc[1]);
    acc[2] = wmma_bf16(a, b2, acc[2]);
    acc[3] = wmma_bf16(a, b3, acc[3]);

    __shared__ float red[16];
    if (threadIdx.x < 16) red[threadIdx.x] = 0.f;
    __syncthreads();

    float part[8];
#pragma unroll
    for (int j = 0; j < 8; ++j) part[j] = 0.f;
#pragma unroll
    for (int j = 0; j < 8; ++j) {
        int rj = r0 + j + 8 * half;
        if (rj < rows) {
            int rbj = rj / LN; int rem = rj % LN; int pj = rem / N; int nj = rem % N;
            int rhc = cell_off(level - nj - 1) + pj + nj + 1;
            const float* rrow = chart_h + ((size_t)rbj * NCELLS + rhc) * SS;
            float p = 0.f;
#pragma unroll
            for (int nt = 0; nt < 4; ++nt)
                p += acc[nt][j] * rrow[colbase + nt * 16 + q];
            part[j] = p;
        }
    }
#pragma unroll
    for (int j = 0; j < 8; ++j) {
        float v = part[j];
        v += __shfl_xor(v, 1); v += __shfl_xor(v, 2);
        v += __shfl_xor(v, 4); v += __shfl_xor(v, 8);
        part[j] = v;
    }
    if (q == 0) {
#pragma unroll
        for (int j = 0; j < 8; ++j) atomicAdd(&red[j + 8 * half], part[j]);
    }
    __syncthreads();

    if (threadIdx.x < 16) {
        int rj = r0 + threadIdx.x;
        if (rj < rows) {
            int rbj = rj / LN; int rem = rj % LN; int pj = rem / N; int nj = rem % N;
            int lhcj = cell_off(nj) + pj;
            int rhcj = cell_off(level - nj - 1) + pj + nj + 1;
            sbuf[rj] = red[threadIdx.x]
                     + chart_s[(size_t)rbj * NCELLS + lhcj]
                     + chart_s[(size_t)rbj * NCELLS + rhcj];
        }
    }
}

// ---------------- softmax over splits, per cell ----------------
__global__ __launch_bounds__(32)
void softmax_kernel(const float* __restrict__ sbuf, float* __restrict__ pbuf,
                    float* __restrict__ chart_s, int level, int L) {
    const int cellidx = blockIdx.x;           // b*L + pos
    const int b = cellidx / L, pos = cellidx % L;
    const int N = level;
    const int lane = threadIdx.x;
    size_t base = (size_t)cellidx * N;

    float s = (lane < N) ? sbuf[base + lane] : -1e30f;
    float m = s;
#pragma unroll
    for (int mask = 16; mask; mask >>= 1) m = fmaxf(m, __shfl_xor(m, mask));
    float e = (lane < N) ? expf(s - m) : 0.f;
    float sum = e;
#pragma unroll
    for (int mask = 16; mask; mask >>= 1) sum += __shfl_xor(sum, mask);
    float p = e / sum;
    if (lane < N) pbuf[base + lane] = p;
    float sp = (lane < N) ? s * p : 0.f;
#pragma unroll
    for (int mask = 16; mask; mask >>= 1) sp += __shfl_xor(sp, mask);
    if (lane == 0)
        chart_s[(size_t)b * NCELLS + cell_off(level) + pos] = sp;
}

// ---------------- compose: hbar = unit(sum_n p_n relu([lh,rh] @ Wc + b)) ----
// grid = B*L blocks (one per output cell), 256 threads
__global__ __launch_bounds__(256)
void compose_kernel(const __bf16* chart_hb, const __bf16* __restrict__ wT,
                    const float* __restrict__ bias, const float* __restrict__ pbuf,
                    float* chart_h, __bf16* chart_hb_out, int level, int L) {
    const int lane = threadIdx.x & 31;
    const int wave = threadIdx.x >> 5;
    const int half = lane >> 4;
    const int q    = lane & 15;
    const int cellidx = blockIdx.x;
    const int b = cellidx / L, pos = cellidx % L;
    const int N = level;
    const bool two = (N > 16);
    const int colbase = wave * 64;

    __shared__ float ssum;
    if (threadIdx.x == 0) ssum = 0.f;

    const __bf16* lhrow[2];
    const __bf16* rhrow[2];
#pragma unroll
    for (int mt = 0; mt < 2; ++mt) {
        int n  = mt * 16 + q;
        int nn = (n < N) ? n : 0;
        int lhc = cell_off(nn) + pos;
        int rhc = cell_off(level - nn - 1) + pos + nn + 1;
        lhrow[mt] = chart_hb + ((size_t)b * NCELLS + lhc) * SS;
        rhrow[mt] = chart_hb + ((size_t)b * NCELLS + rhc) * SS;
    }
    const __bf16* bbase = wT + (size_t)(colbase + q) * DD;

    v8f acc[2][4] = {};
    // pipeline: preload K-step 0
    v16bf a0 = load_a_tile(lhrow[0], half);
    v16bf a1 = {};
    if (two) a1 = load_a_tile(lhrow[1], half);
    v16bf b0 = load_b_tile(bbase + 0 * 16 * DD, half);
    v16bf b1 = load_b_tile(bbase + 1 * 16 * DD, half);
    v16bf b2 = load_b_tile(bbase + 2 * 16 * DD, half);
    v16bf b3 = load_b_tile(bbase + 3 * 16 * DD, half);

    for (int kb = 32; kb < 2 * SS; kb += 32) {
        const __bf16* s0 = (kb < SS) ? (lhrow[0] + kb) : (rhrow[0] + (kb - SS));
        v16bf an0 = load_a_tile(s0, half);
        v16bf an1 = {};
        if (two) {
            const __bf16* s1 = (kb < SS) ? (lhrow[1] + kb) : (rhrow[1] + (kb - SS));
            an1 = load_a_tile(s1, half);
        }
        v16bf bn0 = load_b_tile(bbase + 0 * 16 * DD + kb, half);
        v16bf bn1 = load_b_tile(bbase + 1 * 16 * DD + kb, half);
        v16bf bn2 = load_b_tile(bbase + 2 * 16 * DD + kb, half);
        v16bf bn3 = load_b_tile(bbase + 3 * 16 * DD + kb, half);

        acc[0][0] = wmma_bf16(a0, b0, acc[0][0]);
        acc[0][1] = wmma_bf16(a0, b1, acc[0][1]);
        acc[0][2] = wmma_bf16(a0, b2, acc[0][2]);
        acc[0][3] = wmma_bf16(a0, b3, acc[0][3]);
        if (two) {
            acc[1][0] = wmma_bf16(a1, b0, acc[1][0]);
            acc[1][1] = wmma_bf16(a1, b1, acc[1][1]);
            acc[1][2] = wmma_bf16(a1, b2, acc[1][2]);
            acc[1][3] = wmma_bf16(a1, b3, acc[1][3]);
        }
        a0 = an0; a1 = an1; b0 = bn0; b1 = bn1; b2 = bn2; b3 = bn3;
    }
    acc[0][0] = wmma_bf16(a0, b0, acc[0][0]);
    acc[0][1] = wmma_bf16(a0, b1, acc[0][1]);
    acc[0][2] = wmma_bf16(a0, b2, acc[0][2]);
    acc[0][3] = wmma_bf16(a0, b3, acc[0][3]);
    if (two) {
        acc[1][0] = wmma_bf16(a1, b0, acc[1][0]);
        acc[1][1] = wmma_bf16(a1, b1, acc[1][1]);
        acc[1][2] = wmma_bf16(a1, b2, acc[1][2]);
        acc[1][3] = wmma_bf16(a1, b3, acc[1][3]);
    }

    size_t pbase = (size_t)cellidx * N;
    float bv[4];
#pragma unroll
    for (int nt = 0; nt < 4; ++nt) bv[nt] = bias[colbase + nt * 16 + q];

    float hbar[4] = {0.f, 0.f, 0.f, 0.f};
#pragma unroll
    for (int mt = 0; mt < 2; ++mt) {
        if (mt == 0 || two) {
#pragma unroll
            for (int j = 0; j < 8; ++j) {
                int n = mt * 16 + j + 8 * half;
                float pv = (n < N) ? pbuf[pbase + n] : 0.f;
#pragma unroll
                for (int nt = 0; nt < 4; ++nt) {
                    float h = acc[mt][nt][j] + bv[nt];
                    h = h > 0.f ? h : 0.f;
                    hbar[nt] += pv * h;
                }
            }
        }
    }
    // lane l and l^16 hold the same column, complementary rows
#pragma unroll
    for (int nt = 0; nt < 4; ++nt) hbar[nt] += __shfl_xor(hbar[nt], 16);

    float ssl = 0.f;
#pragma unroll
    for (int nt = 0; nt < 4; ++nt) ssl += hbar[nt] * hbar[nt];
    ssl += __shfl_xor(ssl, 1); ssl += __shfl_xor(ssl, 2);
    ssl += __shfl_xor(ssl, 4); ssl += __shfl_xor(ssl, 8);

    __syncthreads();
    if (lane == 0) atomicAdd(&ssum, ssl);
    __syncthreads();

    float inv = 1.f / fmaxf(sqrtf(ssum), 1e-12f);
    if (half == 0) {
        int outc = cell_off(level) + pos;
        size_t base = ((size_t)b * NCELLS + outc) * SS;
#pragma unroll
        for (int nt = 0; nt < 4; ++nt) {
            int col = colbase + nt * 16 + q;
            float v = hbar[nt] * inv;
            chart_h[base + col]      = v;
            chart_hb_out[base + col] = f2bf(v);
        }
    }
}

// ---------------- host ----------------

extern "C" void kernel_launch(void* const* d_in, const int* in_sizes, int n_in,
                              void* d_out, int out_size, void* d_ws, size_t ws_size,
                              hipStream_t stream) {
    const float* x      = (const float*)d_in[0];
    const float* w_leaf = (const float*)d_in[1];
    const float* b_leaf = (const float*)d_in[2];
    const float* w_comp = (const float*)d_in[3];
    const float* b_comp = (const float*)d_in[4];
    const float* s_bil  = (const float*)d_in[5];
    float* chart_h = (float*)d_out;

    char* ws = (char*)d_ws;
    auto align256 = [](size_t v) { return (v + 255) & ~(size_t)255; };
    size_t o = 0;
    __bf16* chart_hb = (__bf16*)(ws + o); o = align256(o + (size_t)BB * NCELLS * SS * 2);
    __bf16* xb       = (__bf16*)(ws + o); o = align256(o + (size_t)BB * TT * DD * 2);
    __bf16* wTleaf   = (__bf16*)(ws + o); o = align256(o + (size_t)SS * DD * 2);
    __bf16* wTcomp   = (__bf16*)(ws + o); o = align256(o + (size_t)SS * DD * 2);
    __bf16* sT       = (__bf16*)(ws + o); o = align256(o + (size_t)SS * SS * 2);
    float*  chart_s  = (float*)(ws + o);  o = align256(o + (size_t)BB * NCELLS * 4);
    float*  sbuf     = (float*)(ws + o);  o = align256(o + (size_t)BB * 256 * 4);
    float*  pbuf     = (float*)(ws + o);  o = align256(o + (size_t)BB * 256 * 4);
    (void)ws_size; (void)in_sizes; (void)n_in; (void)out_size;

    int nx = BB * TT * DD;
    cvt_x_kernel<<<(nx + 255) / 256, 256, 0, stream>>>(x, xb, nx);
    transpose_w_kernel<<<(SS * DD + 255) / 256, 256, 0, stream>>>(w_leaf, wTleaf, DD, SS);
    transpose_w_kernel<<<(SS * DD + 255) / 256, 256, 0, stream>>>(w_comp, wTcomp, DD, SS);
    transpose_w_kernel<<<(SS * SS + 255) / 256, 256, 0, stream>>>(s_bil, sT, SS, SS);
    zero_f32_kernel<<<(BB * NCELLS + 255) / 256, 256, 0, stream>>>(chart_s, BB * NCELLS);

    leaf_kernel<<<(BB * TT) / 16, 256, 0, stream>>>(xb, wTleaf, b_leaf, chart_h, chart_hb);

    for (int level = 1; level < TT; ++level) {
        int L = TT - level, N = level;
        int rows = BB * L * N;
        score_kernel<<<(rows + 15) / 16, 256, 0, stream>>>(
            chart_hb, chart_h, sT, chart_s, sbuf, level, L, rows);
        softmax_kernel<<<BB * L, 32, 0, stream>>>(sbuf, pbuf, chart_s, level, L);
        compose_kernel<<<BB * L, 256, 0, stream>>>(
            chart_hb, wTcomp, b_comp, pbuf, chart_h, chart_hb, level, L);
    }
}
